// MultiHeadGraphAttention_71236327572065
// MI455X (gfx1250) — matrix operand order repaired and checked by reading, not compiled, not run on hardware
//
#include <hip/hip_runtime.h>
#include <math.h>

// ---------------------------------------------------------------------------
// MultiHeadGraphAttention for MI455X (gfx1250)
//
//   s1[h,n] = hi[n,:] . (W_heads[h] @ a1[h])           (length-256 dot)
//   s2[h,n] = (hj+eij)[n,:] . (W_heads[h] @ a2[h])
//   att[h,i,j] = softmax_j( max(s1[h,i] + s2[h,j], 0) )  (leaky(relu(x)) == relu(x))
//   out = elu( mean_h( att[h] @ (hj@W) ) )
// `adj` (64 MB) is UNUSED by the reference -> never read.
//
// Heavy part (att @ Whj, 8.6 GFLOP): v_wmma_f32_16x16x32_f16 with f16 attention
// tiles built in LDS (f32 exp -> f16, denominator accumulated from the SAME
// rounded weights so numerator/denominator are consistent) and Whj pre-packed
// f16-transposed so B fragments are contiguous b128 LDS loads.
// Whj itself is computed in full fp32 via v_wmma_f32_16x16x4_f32.
// ---------------------------------------------------------------------------

#define GN    4096
#define GFIN  256
#define GFOUT 64
#define GH    4
#define JT    128          // j-tile width in the fused attention kernel
#define WP    (JT + 8)     // f16 LDS row stride (136 halves = 272B, 16B-aligned rows)
#define ASTR  (GFIN + 4)   // fp32 LDS stride for hj tile in Whj GEMM

typedef __attribute__((ext_vector_type(2)))  float    v2f;
typedef __attribute__((ext_vector_type(8)))  float    v8f;
typedef __attribute__((ext_vector_type(8)))  _Float16 v8h;
typedef __attribute__((ext_vector_type(16))) _Float16 v16h;

// ---- Kernel 1: per-head projection vectors q1[h][k], q2[h][k] --------------
__global__ void proj_vec_kernel(const float* __restrict__ W_heads,
                                const float* __restrict__ a_heads,
                                float* __restrict__ q1, float* __restrict__ q2) {
  int t = blockIdx.x * blockDim.x + threadIdx.x;   // 0 .. GH*GFIN-1
  int h = t / GFIN, k = t % GFIN;
  const float* wrow = W_heads + ((size_t)h * GFIN + k) * GFOUT;
  const float* a1 = a_heads + (size_t)h * 2 * GFOUT;
  const float* a2 = a1 + GFOUT;
  float r1 = 0.f, r2 = 0.f;
  #pragma unroll 8
  for (int o = 0; o < GFOUT; ++o) {
    float w = wrow[o];
    r1 = fmaf(w, a1[o], r1);
    r2 = fmaf(w, a2[o], r2);
  }
  q1[t] = r1;
  q2[t] = r2;
}

// ---- Kernel 2: attention scores s1[h][n], s2[h][n] -------------------------
__global__ void score_kernel(const float* __restrict__ hi,
                             const float* __restrict__ hj,
                             const float* __restrict__ eij,
                             const float* __restrict__ q1,
                             const float* __restrict__ q2,
                             float* __restrict__ s1, float* __restrict__ s2) {
  int t = blockIdx.x * blockDim.x + threadIdx.x;   // 0 .. GH*GN-1
  int h = t / GN, n = t % GN;
  const float4* xi = (const float4*)(hi  + (size_t)n * GFIN);
  const float4* xj = (const float4*)(hj  + (size_t)n * GFIN);
  const float4* xe = (const float4*)(eij + (size_t)n * GFIN);
  const float4* p1 = (const float4*)(q1 + (size_t)h * GFIN);
  const float4* p2 = (const float4*)(q2 + (size_t)h * GFIN);
  float a = 0.f, b = 0.f;
  #pragma unroll 4
  for (int k = 0; k < GFIN / 4; ++k) {
    float4 vi = xi[k], vj = xj[k], ve = xe[k], w1 = p1[k], w2 = p2[k];
    a = fmaf(vi.x, w1.x, fmaf(vi.y, w1.y, fmaf(vi.z, w1.z, fmaf(vi.w, w1.w, a))));
    float sx = vj.x + ve.x, sy = vj.y + ve.y, sz = vj.z + ve.z, sw = vj.w + ve.w;
    b = fmaf(sx, w2.x, fmaf(sy, w2.y, fmaf(sz, w2.z, fmaf(sw, w2.w, b))));
  }
  s1[t] = a;
  s2[t] = b;
}

// ---- Kernel 3: per-head max of s2 (for stable softmax) ---------------------
__global__ void s2max_kernel(const float* __restrict__ s2, float* __restrict__ s2max) {
  __shared__ float red[256];
  int h = blockIdx.x;
  float m = -3.0e38f;
  for (int n = threadIdx.x; n < GN; n += 256) m = fmaxf(m, s2[(size_t)h * GN + n]);
  red[threadIdx.x] = m;
  __syncthreads();
  for (int s = 128; s > 0; s >>= 1) {
    if (threadIdx.x < s) red[threadIdx.x] = fmaxf(red[threadIdx.x], red[threadIdx.x + s]);
    __syncthreads();
  }
  if (threadIdx.x == 0) s2max[h] = red[0];
}

// ---- Kernel 4: Whj = hj @ W (fp32 WMMA 16x16x4) -> f16-transposed whjT -----
// whjT layout: [o][j], o = 0..63, j = 0..4095 -- exactly the contiguous-K
// per-lane layout this code's f16 WMMA B fragment loads use.
__global__ __launch_bounds__(128)
void whj_gemm_kernel(const float* __restrict__ hj, const float* __restrict__ W,
                     _Float16* __restrict__ whjT) {
  __shared__ float atile[16][ASTR];     // 16 x 256 hj tile, padded
  int row0 = blockIdx.x * 16;
  int t = threadIdx.x;

  for (int idx = t; idx < 16 * (GFIN / 4); idx += 128) {
    int r = idx / (GFIN / 4), c4 = idx % (GFIN / 4);
    float4 v = ((const float4*)(hj + (size_t)(row0 + r) * GFIN))[c4];
    atile[r][c4 * 4 + 0] = v.x;
    atile[r][c4 * 4 + 1] = v.y;
    atile[r][c4 * 4 + 2] = v.z;
    atile[r][c4 * 4 + 3] = v.w;
  }
  __syncthreads();

  int lane = t & 31, wv = t >> 5;
  int m  = lane & 15;          // A: M index, B: N index
  int kp = lane >> 4;          // K-pair select
  int col = wv * 16 + m;

  v8f acc = {};
  for (int k0 = 0; k0 < GFIN; k0 += 4) {
    int k = k0 + 2 * kp;
    v2f a; a.x = atile[m][k];                a.y = atile[m][k + 1];
    v2f b; b.x = W[(size_t)k * GFOUT + col]; b.y = W[(size_t)(k + 1) * GFOUT + col];
    acc = __builtin_amdgcn_wmma_f32_16x16x4_f32(false, a, false, b, (short)0, acc,
                                                false, false);
  }

  // Lane holds rows M = mbase..mbase+7 of column ncol: contiguous j in whjT
  // -> one 16B store per lane.
  int mbase = (lane < 16) ? 0 : 8;
  int ncol = wv * 16 + (lane & 15);
  v8h o16;
  #pragma unroll
  for (int r = 0; r < 8; ++r) o16[r] = (_Float16)acc[r];
  *(v8h*)(whjT + (size_t)ncol * GN + row0 + mbase) = o16;
}

// ---- Kernel 5: fused attention + aggregation (f16 WMMA 16x16x32) -----------
__global__ __launch_bounds__(128)
void gat_attn_kernel(const float* __restrict__ s1g, const float* __restrict__ s2g,
                     const float* __restrict__ s2maxg,
                     const _Float16* __restrict__ whjT,
                     float* __restrict__ out) {
  __shared__ _Float16 wt[GH][16][WP];  // f16 attention-weight tiles, all heads
  __shared__ _Float16 bt[GFOUT][WP];   // staged whjT tile (64 cols x JT j's)
  __shared__ float s2t[GH][JT];
  __shared__ float s1s[GH][16];
  __shared__ float ms[GH][16];
  __shared__ float dred[GH][16][8];
  __shared__ float denom[GH][16];

  int t = threadIdx.x;
  int row0 = blockIdx.x * 16;

  if (t < GH * 16) {
    int h = t >> 4, r = t & 15;
    float v = s1g[(size_t)h * GN + row0 + r];
    s1s[h][r] = v;
    ms[h][r] = fmaxf(v + s2maxg[h], 0.0f);
  }

  int lane = t & 31, wv = t >> 5;
  int m  = lane & 15;                  // A: M row, B/D: N col (within 16)
  int kp = lane >> 4;                  // half-wave K-chunk select
  int col = wv * 16 + m;               // this wave's output column
  int wr = t >> 3;                     // w-fill row 0..15
  int cg = t & 7;                      // w-fill 16-wide column group
  int bo  = t >> 1;                    // bt-stage: output column 0..63
  int bs  = (t & 1) * (JT / 2);        // bt-stage: half-row offset

  float dpart[GH] = {0.f, 0.f, 0.f, 0.f};
  v8f acc[GH];
  #pragma unroll
  for (int h = 0; h < GH; ++h) acc[h] = (v8f){};

  for (int j0 = 0; j0 < GN; j0 += JT) {
    __syncthreads();                   // prior GEMM reads done before overwrite
    // stage s2 slice
    #pragma unroll
    for (int h = 0; h < GH; ++h) s2t[h][t] = s2g[(size_t)h * GN + j0 + t];
    // stage whjT tile (coalesced 16B chunks; rows are j-contiguous in memory)
    {
      const _Float16* src = whjT + (size_t)bo * GN + j0 + bs;
      _Float16* dst = &bt[bo][bs];
      #pragma unroll
      for (int q = 0; q < (JT / 2) / 8; ++q)
        *(v8h*)(dst + q * 8) = *(const v8h*)(src + q * 8);
      if (j0 + JT < GN)
        __builtin_prefetch(src + JT, 0, 3);   // global_prefetch_b8: next tile
    }
    __syncthreads();

    // Build w = exp(max(s1_i + s2_j, 0) - m_i), round to f16, pack to LDS.
    // Denominator accumulated from the ROUNDED value -> consistent softmax.
    #pragma unroll
    for (int h = 0; h < GH; ++h) {
      float s1v = s1s[h][wr];
      float mv  = ms[h][wr];
      float d = 0.f;
      v8h pk0, pk1;
      #pragma unroll
      for (int c = 0; c < 16; ++c) {
        int j = cg * 16 + c;
        float wf = __expf(fmaxf(s1v + s2t[h][j], 0.0f) - mv);
        _Float16 wh = (_Float16)wf;
        d += (float)wh;
        if (c < 8) pk0[c] = wh; else pk1[c - 8] = wh;
      }
      *(v8h*)(&wt[h][wr][cg * 16])     = pk0;   // ds_store_b128
      *(v8h*)(&wt[h][wr][cg * 16 + 8]) = pk1;
      dpart[h] += d;
    }
    __syncthreads();

    // numer[h] += w[h](16xJT) @ Whj(JTx64), f16 WMMA, K=32 per instruction.
    #pragma unroll
    for (int kw = 0; kw < JT; kw += 32) {
      // B fragment: lane = column `col`, 16 contiguous K at kw + 16*kp
      const _Float16* brow = &bt[col][kw + 16 * kp];
      v8h b0 = *(const v8h*)brow;
      v8h b1 = *(const v8h*)(brow + 8);
      v16h b;
      #pragma unroll
      for (int q = 0; q < 8; ++q) { b[q] = b0[q]; b[8 + q] = b1[q]; }
      #pragma unroll
      for (int h = 0; h < GH; ++h) {
        // A fragment: row m, 16 contiguous K at kw + 16*kp
        const _Float16* arow = &wt[h][m][kw + 16 * kp];
        v8h a0 = *(const v8h*)arow;
        v8h a1 = *(const v8h*)(arow + 8);
        v16h a;
        #pragma unroll
        for (int q = 0; q < 8; ++q) { a[q] = a0[q]; a[8 + q] = a1[q]; }
        acc[h] = __builtin_amdgcn_wmma_f32_16x16x32_f16(false, a, false, b, (short)0,
                                                        acc[h], false, false);
      }
    }
  }

  // reduce softmax denominators
  __syncthreads();
  #pragma unroll
  for (int h = 0; h < GH; ++h) dred[h][wr][cg] = dpart[h];
  __syncthreads();
  if (t < GH * 16) {
    int h = t >> 4, r = t & 15;
    float d = 0.f;
    #pragma unroll
    for (int c = 0; c < 8; ++c) d += dred[h][r][c];
    denom[h][r] = d;
  }
  __syncthreads();

  // epilogue: normalize, mean over heads, ELU, store
  int mbase = (lane < 16) ? 0 : 8;
  int ncol = wv * 16 + (lane & 15);
  #pragma unroll
  for (int r = 0; r < 8; ++r) {
    int M = mbase + r;
    float s = 0.f;
    #pragma unroll
    for (int h = 0; h < GH; ++h) s += acc[h][r] / denom[h][M];
    s *= 0.25f;
    float o = (s > 0.0f) ? s : (__expf(s) - 1.0f);   // elu, alpha=1
    out[(size_t)(row0 + M) * GFOUT + ncol] = o;
  }
}

// ---------------------------------------------------------------------------
extern "C" void kernel_launch(void* const* d_in, const int* in_sizes, int n_in,
                              void* d_out, int out_size, void* d_ws, size_t ws_size,
                              hipStream_t stream) {
  (void)in_sizes; (void)n_in; (void)out_size; (void)ws_size;
  const float* hi      = (const float*)d_in[0];
  const float* hj      = (const float*)d_in[1];
  const float* eij     = (const float*)d_in[2];
  // d_in[3] = adj : UNUSED by the reference (unmasked softmax) -> never read
  const float* W_heads = (const float*)d_in[4];
  const float* a_heads = (const float*)d_in[5];
  const float* W       = (const float*)d_in[6];
  float* out = (float*)d_out;

  float* ws  = (float*)d_ws;
  float* q1  = ws;                          // GH*GFIN floats
  float* q2  = q1  + GH * GFIN;             // GH*GFIN floats
  float* s1  = q2  + GH * GFIN;             // GH*GN floats
  float* s2  = s1  + (size_t)GH * GN;       // GH*GN floats
  float* s2m = s2  + (size_t)GH * GN;       // 8 floats
  _Float16* whjT = (_Float16*)(s2m + 8);    // GFOUT*GN halves (512 KB)

  proj_vec_kernel<<<(GH * GFIN) / 256, 256, 0, stream>>>(W_heads, a_heads, q1, q2);
  score_kernel<<<(GH * GN) / 256, 256, 0, stream>>>(hi, hj, eij, q1, q2, s1, s2);
  s2max_kernel<<<GH, 256, 0, stream>>>(s2, s2m);
  whj_gemm_kernel<<<GN / 16, 128, 0, stream>>>(hj, W, whjT);
  gat_attn_kernel<<<GN / 16, 128, 0, stream>>>(s1, s2, s2m, whjT, out);
}
